// LINKX_86577950753300
// MI455X (gfx1250) — matrix operand order repaired and checked by reading, not compiled, not run on hardware
//
#include <hip/hip_runtime.h>
#include <hip/hip_bf16.h>

typedef __attribute__((ext_vector_type(16))) __bf16 v16bf;
typedef __attribute__((ext_vector_type(8)))  float  v8f;

#define HIDDIM 128
#define INDIM  256
#define NCLS   40
#define NPADC  48   // 40 padded to 3 x 16-wide WMMA tiles

__device__ __forceinline__ unsigned short f2bf(float f) {
  unsigned int x = __float_as_uint(f);
  x += 0x7FFFu + ((x >> 16) & 1u);   // round-to-nearest-even
  return (unsigned short)(x >> 16);
}
__device__ __forceinline__ float bf2f(unsigned short u) {
  return __uint_as_float(((unsigned int)u) << 16);
}

// ---------- weight transpose + f32->bf16 (with zero column padding) ----------
__global__ void k_wtrans(const float* __restrict__ w, unsigned short* __restrict__ wt,
                         int K, int Nn, int Npad) {
  int idx = blockIdx.x * blockDim.x + threadIdx.x;
  if (idx >= K * Npad) return;
  int n = idx % Npad;
  int k = idx / Npad;
  wt[(size_t)n * K + k] = (n < Nn) ? f2bf(w[(size_t)k * Nn + n]) : (unsigned short)0;
}

// ---------- edge scatter: one wave32 per edge, 4 columns per lane ----------
__global__ __launch_bounds__(256) void k_scatter(const int* __restrict__ ei,
                                                 const float* __restrict__ emb,
                                                 float* __restrict__ agg,
                                                 float* __restrict__ deg, int E) {
  int lane = threadIdx.x & 31;
  int wave = threadIdx.x >> 5;
  long long e = (long long)blockIdx.x * 8 + wave;
  if (e >= E) return;
  int src = ei[e];
  int dst = ei[(long long)E + e];
  const float* s = emb + (size_t)src * HIDDIM;
  float*       d = agg + (size_t)dst * HIDDIM;
#pragma unroll
  for (int i = 0; i < 4; ++i) {
    int c = lane + 32 * i;
    atomicAdd(&d[c], s[c]);        // global_atomic_add_f32, L2-resident target
  }
  if (lane == 0) atomicAdd(&deg[dst], 1.0f);
}

// ---------- mean-normalize (deg >= 1) and convert to bf16 ----------
__global__ void k_norm(const float* __restrict__ agg, const float* __restrict__ deg,
                       unsigned short* __restrict__ o, int total) {
  int idx = blockIdx.x * blockDim.x + threadIdx.x;
  if (idx >= total) return;
  float dg = deg[idx >> 7];
  dg = dg < 1.0f ? 1.0f : dg;
  o[idx] = f2bf(agg[idx] / dg);
}

// ---------- WMMA GEMM: C[M,Nn] = epilogue(A[M,KT] @ Bt^T + bias) ----------
// AMODE: 0 = A bf16 [M,KT]; 1 = A f32 [M,KT] (bf16 on the fly);
//        2 = A = concat of two bf16 [M,HIDDIM] halves (Ap | Ap2), KT = 256.
// Bt: pre-transposed bf16 [Npad, KT] (row = output column, contiguous K).
// NT column tiles per wave (A fragment reused NT times). K loop fully unrolled.
template <int AMODE, int KT, int NT, bool RELU, bool RESID, bool OUTF32, bool COLGUARD>
__global__ __launch_bounds__(256) void k_gemm(const void* __restrict__ Ap,
                                              const void* __restrict__ Ap2,
                                              const unsigned short* __restrict__ Bt,
                                              const float* __restrict__ bias,
                                              const unsigned short* __restrict__ r1,
                                              const unsigned short* __restrict__ r2,
                                              void* __restrict__ Out,
                                              int M, int Nn) {
  const int lane = threadIdx.x & 31;
  const int wave = threadIdx.x >> 5;
  const int m0 = (blockIdx.x * 8 + wave) * 16;   // 8 waves stacked along M
  if (m0 >= M) return;                            // whole-wave exit: EXEC all-ones for WMMA
  const int n0 = blockIdx.y * (16 * NT);
  const int rs = lane & 15;                       // row-in-tile (A) / col-in-tile (B,C)
  const int hh = lane >> 4;                       // K-half select per ISA layout
  int rA = m0 + rs;
  if (rA > M - 1) rA = M - 1;                     // clamp loads; stores handled below

  v8f acc[NT];
#pragma unroll
  for (int t = 0; t < NT; ++t) {
    float bv;
    if (COLGUARD) bv = (n0 + 16 * t + rs < Nn) ? bias[n0 + 16 * t + rs] : 0.0f;
    else          bv = bias[n0 + 16 * t + rs];
#pragma unroll
    for (int v = 0; v < 8; ++v) acc[t][v] = bv;   // D = A*B + (bias-seeded C)
  }

  union Frag { unsigned short s[16]; uint4 q[2]; v16bf v; };

  const unsigned short* arow_bf = (AMODE == 0) ? (const unsigned short*)Ap + (size_t)rA * KT
                                               : (const unsigned short*)nullptr;
  const float*          arow_f  = (AMODE == 1) ? (const float*)Ap + (size_t)rA * KT
                                               : (const float*)nullptr;

#pragma unroll
  for (int kk = 0; kk < KT; kk += 32) {
    Frag ua;
    // A fragment: lanes 0-15 row M, K chunks {8hh..8hh+7} and {16+8hh..16+8hh+7}
    const int ka0 = kk + hh * 8;
    const int ka1 = ka0 + 16;
    if (AMODE == 0) {
      ua.q[0] = *(const uint4*)(arow_bf + ka0);
      ua.q[1] = *(const uint4*)(arow_bf + ka1);
    } else if (AMODE == 1) {
      float4 f0 = *(const float4*)(arow_f + ka0);
      float4 f1 = *(const float4*)(arow_f + ka0 + 4);
      float4 f2 = *(const float4*)(arow_f + ka1);
      float4 f3 = *(const float4*)(arow_f + ka1 + 4);
      ua.s[0]  = f2bf(f0.x); ua.s[1]  = f2bf(f0.y); ua.s[2]  = f2bf(f0.z); ua.s[3]  = f2bf(f0.w);
      ua.s[4]  = f2bf(f1.x); ua.s[5]  = f2bf(f1.y); ua.s[6]  = f2bf(f1.z); ua.s[7]  = f2bf(f1.w);
      ua.s[8]  = f2bf(f2.x); ua.s[9]  = f2bf(f2.y); ua.s[10] = f2bf(f2.z); ua.s[11] = f2bf(f2.w);
      ua.s[12] = f2bf(f3.x); ua.s[13] = f2bf(f3.y); ua.s[14] = f2bf(f3.z); ua.s[15] = f2bf(f3.w);
    } else {  // concat: 8-wide chunks never straddle the 128 boundary (resolved at unroll time)
      const unsigned short* p0 = (ka0 < HIDDIM)
          ? (const unsigned short*)Ap  + (size_t)rA * HIDDIM + ka0
          : (const unsigned short*)Ap2 + (size_t)rA * HIDDIM + (ka0 - HIDDIM);
      const unsigned short* p1 = (ka1 < HIDDIM)
          ? (const unsigned short*)Ap  + (size_t)rA * HIDDIM + ka1
          : (const unsigned short*)Ap2 + (size_t)rA * HIDDIM + (ka1 - HIDDIM);
      ua.q[0] = *(const uint4*)p0;
      ua.q[1] = *(const uint4*)p1;
    }
    // B fragments: lane col N, K chunks {8hh..8hh+7} and {16+8hh..16+8hh+7}
    const int kb = kk + hh * 8;
#pragma unroll
    for (int t = 0; t < NT; ++t) {
      const unsigned short* btrow = Bt + (size_t)(n0 + 16 * t + rs) * KT;
      Frag ub;
      ub.q[0] = *(const uint4*)(btrow + kb);
      ub.q[1] = *(const uint4*)(btrow + kb + 16);
      acc[t] = __builtin_amdgcn_wmma_f32_16x16x32_bf16(false, ua.v, false, ub.v,
                                                       (short)0, acc[t], false, false);
    }
  }

  // C/D layout: VGPR v -> row m0 + v + 8*hh, col = tile col + (lane&15)
  const bool fullM = (m0 + 16 <= M);   // always true when 16 | M (M=100000)
#pragma unroll
  for (int t = 0; t < NT; ++t) {
    const int c = n0 + 16 * t + rs;
    if (COLGUARD && c >= Nn) continue;
    if (fullM) {
#pragma unroll
      for (int v = 0; v < 8; ++v) {
        const int r = m0 + v + 8 * hh;
        float val = acc[t][v];
        if (RELU)  val = fmaxf(val, 0.0f);
        if (RESID) val = fmaxf(val + bf2f(r1[(size_t)r * HIDDIM + c])
                                   + bf2f(r2[(size_t)r * HIDDIM + c]), 0.0f);
        if (OUTF32) ((float*)Out)[(size_t)r * Nn + c] = val;
        else        ((unsigned short*)Out)[(size_t)r * Nn + c] = f2bf(val);
      }
    } else {
#pragma unroll
      for (int v = 0; v < 8; ++v) {
        const int r = m0 + v + 8 * hh;
        if (r < M) {
          float val = acc[t][v];
          if (RELU)  val = fmaxf(val, 0.0f);
          if (RESID) val = fmaxf(val + bf2f(r1[(size_t)r * HIDDIM + c])
                                     + bf2f(r2[(size_t)r * HIDDIM + c]), 0.0f);
          if (OUTF32) ((float*)Out)[(size_t)r * Nn + c] = val;
          else        ((unsigned short*)Out)[(size_t)r * Nn + c] = f2bf(val);
        }
      }
    }
  }
}

extern "C" void kernel_launch(void* const* d_in, const int* in_sizes, int n_in,
                              void* d_out, int out_size, void* d_ws, size_t ws_size,
                              hipStream_t stream) {
  const float* x   = (const float*)d_in[0];
  const int*   ei  = (const int*)d_in[1];
  const float* emb = (const float*)d_in[2];
  const float* wx1 = (const float*)d_in[3];
  const float* bx1 = (const float*)d_in[4];
  const float* wx2 = (const float*)d_in[5];
  const float* bx2 = (const float*)d_in[6];
  const float* wa1 = (const float*)d_in[7];
  const float* ba1 = (const float*)d_in[8];
  const float* wa2 = (const float*)d_in[9];
  const float* ba2 = (const float*)d_in[10];
  const float* ww1 = (const float*)d_in[11];
  const float* bw1 = (const float*)d_in[12];
  const float* wc  = (const float*)d_in[13];
  const float* bc  = (const float*)d_in[14];

  const int N = in_sizes[0] / INDIM;  // 100000
  const int E = in_sizes[1] / 2;      // 1600000

  char* base = (char*)d_ws;
  size_t off = 0;
  auto grab = [&](size_t bytes) -> void* {
    void* p = base + off;
    off = (off + bytes + 255) & ~((size_t)255);
    return p;
  };
  float*          agg   = (float*)grab((size_t)N * HIDDIM * sizeof(float));
  float*          deg   = (float*)grab((size_t)N * sizeof(float));
  unsigned short* haraw = (unsigned short*)grab((size_t)N * HIDDIM * 2);
  unsigned short* tmp   = (unsigned short*)grab((size_t)N * HIDDIM * 2);  // hx1 / ha1 / h
  unsigned short* hx    = (unsigned short*)grab((size_t)N * HIDDIM * 2);
  unsigned short* ha    = (unsigned short*)grab((size_t)N * HIDDIM * 2);
  unsigned short* wx1t  = (unsigned short*)grab((size_t)INDIM * HIDDIM * 2);
  unsigned short* wx2t  = (unsigned short*)grab((size_t)HIDDIM * HIDDIM * 2);
  unsigned short* wa1t  = (unsigned short*)grab((size_t)HIDDIM * HIDDIM * 2);
  unsigned short* wa2t  = (unsigned short*)grab((size_t)HIDDIM * HIDDIM * 2);
  unsigned short* ww1t  = (unsigned short*)grab((size_t)2 * HIDDIM * HIDDIM * 2);
  unsigned short* wct   = (unsigned short*)grab((size_t)NPADC * HIDDIM * 2);

  hipMemsetAsync(agg, 0, (size_t)N * HIDDIM * sizeof(float), stream);
  hipMemsetAsync(deg, 0, (size_t)N * sizeof(float), stream);

  k_wtrans<<<(INDIM * HIDDIM + 255) / 256, 256, 0, stream>>>(wx1, wx1t, INDIM, HIDDIM, HIDDIM);
  k_wtrans<<<(HIDDIM * HIDDIM + 255) / 256, 256, 0, stream>>>(wx2, wx2t, HIDDIM, HIDDIM, HIDDIM);
  k_wtrans<<<(HIDDIM * HIDDIM + 255) / 256, 256, 0, stream>>>(wa1, wa1t, HIDDIM, HIDDIM, HIDDIM);
  k_wtrans<<<(HIDDIM * HIDDIM + 255) / 256, 256, 0, stream>>>(wa2, wa2t, HIDDIM, HIDDIM, HIDDIM);
  k_wtrans<<<(2 * HIDDIM * HIDDIM + 255) / 256, 256, 0, stream>>>(ww1, ww1t, 2 * HIDDIM, HIDDIM, HIDDIM);
  k_wtrans<<<(HIDDIM * NPADC + 255) / 256, 256, 0, stream>>>(wc, wct, HIDDIM, NCLS, NPADC);

  k_scatter<<<(E + 7) / 8, 256, 0, stream>>>(ei, emb, agg, deg, E);
  k_norm<<<(N * HIDDIM + 255) / 256, 256, 0, stream>>>(agg, deg, haraw, N * HIDDIM);

  const int gx = (N + 127) / 128;   // 128-row block tiles
  dim3 blk(256);
  // hx1 = relu(x @ wx1 + bx1)               [f32 A, on-the-fly bf16]
  k_gemm<1, INDIM, 4, true, false, false, false><<<dim3(gx, 2), blk, 0, stream>>>(
      x, nullptr, wx1t, bx1, nullptr, nullptr, tmp, N, HIDDIM);
  // hx = hx1 @ wx2 + bx2
  k_gemm<0, HIDDIM, 4, false, false, false, false><<<dim3(gx, 2), blk, 0, stream>>>(
      tmp, nullptr, wx2t, bx2, nullptr, nullptr, hx, N, HIDDIM);
  // ha1 = relu(haraw @ wa1 + ba1)
  k_gemm<0, HIDDIM, 4, true, false, false, false><<<dim3(gx, 2), blk, 0, stream>>>(
      haraw, nullptr, wa1t, ba1, nullptr, nullptr, tmp, N, HIDDIM);
  // ha = ha1 @ wa2 + ba2
  k_gemm<0, HIDDIM, 4, false, false, false, false><<<dim3(gx, 2), blk, 0, stream>>>(
      tmp, nullptr, wa2t, ba2, nullptr, nullptr, ha, N, HIDDIM);
  // h = relu(relu(concat(hx,ha) @ ww1 + bw1) + hx + ha)
  k_gemm<2, 2 * HIDDIM, 4, true, true, false, false><<<dim3(gx, 2), blk, 0, stream>>>(
      hx, ha, ww1t, bw1, hx, ha, tmp, N, HIDDIM);
  // logits = h @ wc + bc   (f32 out, 48 padded cols in one pass, masked to 40)
  k_gemm<0, HIDDIM, 3, false, false, true, true><<<dim3(gx, 1), blk, 0, stream>>>(
      tmp, nullptr, wct, bc, nullptr, nullptr, d_out, N, NCLS);

  (void)n_in; (void)out_size; (void)ws_size;
}
